// VectorQuantize_31636729102595
// MI455X (gfx1250) — matrix operand order repaired and checked by reading, not compiled, not run on hardware
//
#include <hip/hip_runtime.h>
#include <hip/hip_bf16.h>

// ---------------------------------------------------------------------------
// VQ codebook search on gfx1250 (MI455X):
//   scores = z @ cb^T via bf16 "hi+lo" split (3x v_wmma_f32_16x16x32_bf16 per
//   fp32 tile-step: Ahi*Bhi + Ahi*Blo + Alo*Bhi), fp32 accumulate.
//   4 independent accumulator chains per wave to keep the XDL pipe busy.
//   argmin(dist) == argmax(2*dot - ||e||^2); tracked as packed u64 keys.
// ---------------------------------------------------------------------------

typedef __attribute__((ext_vector_type(16))) __bf16 v16bf;
typedef __attribute__((ext_vector_type(8)))  float  v8f;

#define DIMD   512
#define KCB    8192
#define NTOK   32768
#define MTILE  64
#define NTILE  128   // codes per chunk (2 code-groups x 4 tiles x 16)
#define LDST   520   // padded LDS row stride (bf16 elems) -> conflict-free b128 loads

union FragB16 { uint4 q[2]; v16bf v; };

__device__ __forceinline__ unsigned long long packKey(float s, int code) {
    unsigned u = __float_as_uint(s);
    u = (u & 0x80000000u) ? ~u : (u | 0x80000000u);   // order-preserving float->uint
    // tie-break: larger low word == smaller code  (matches argmin "first" rule)
    return ((unsigned long long)u << 32) | (unsigned)(8191 - code);
}

__device__ __forceinline__ unsigned long long shflXor64(unsigned long long v, int m) {
    unsigned lo = (unsigned)v, hi = (unsigned)(v >> 32);
    lo = __shfl_xor(lo, m, 32);
    hi = __shfl_xor(hi, m, 32);
    return ((unsigned long long)hi << 32) | lo;
}

// ---- prep: split fp32 codebook into bf16 hi/lo ----------------------------
__global__ __launch_bounds__(256)
void cb_split_kernel(const float* __restrict__ cb,
                     __bf16* __restrict__ hi, __bf16* __restrict__ lo) {
    int i = blockIdx.x * 256 + threadIdx.x;       // one float4 per thread
    float4 v = ((const float4*)cb)[i];
    int o = i * 4;
    __bf16 h0 = (__bf16)v.x, h1 = (__bf16)v.y, h2 = (__bf16)v.z, h3 = (__bf16)v.w;
    hi[o+0] = h0; hi[o+1] = h1; hi[o+2] = h2; hi[o+3] = h3;
    lo[o+0] = (__bf16)(v.x - (float)h0);
    lo[o+1] = (__bf16)(v.y - (float)h1);
    lo[o+2] = (__bf16)(v.z - (float)h2);
    lo[o+3] = (__bf16)(v.w - (float)h3);
}

// ---- prep: ||e_k||^2 in fp32, one wave per codebook row --------------------
__global__ __launch_bounds__(256)
void cb_norms_kernel(const float* __restrict__ cb, float* __restrict__ norms) {
    int row  = blockIdx.x * 8 + (threadIdx.x >> 5);
    int lane = threadIdx.x & 31;
    const float* r = cb + (size_t)row * DIMD;
    float s = 0.0f;
    #pragma unroll
    for (int j = 0; j < DIMD / 32; ++j) {
        float e = r[lane + 32 * j];
        s = fmaf(e, e, s);
    }
    #pragma unroll
    for (int m = 16; m >= 1; m >>= 1) s += __shfl_xor(s, m, 32);
    if (lane == 0) norms[row] = s;
}

// ---- main: tiled bf16x3 GEMM + fused argmax --------------------------------
__global__ __launch_bounds__(256)
void vq_argmin_kernel(const float* __restrict__ z,
                      const __bf16* __restrict__ cb_hi,
                      const __bf16* __restrict__ cb_lo,
                      const float* __restrict__ norms,
                      float* __restrict__ idx_out) {
    __shared__ __bf16 zhi[MTILE * LDST];
    __shared__ __bf16 zlo[MTILE * LDST];
    __shared__ unsigned long long bestL[MTILE];

    const int tid   = threadIdx.x;
    const int wave  = tid >> 5;
    const int lane  = tid & 31;
    const int m16   = lane & 15;
    const int half  = lane >> 4;
    const int strip = wave >> 1;   // 4 row strips of 16 rows
    const int cg    = wave & 1;    // 2 code groups of 64 codes

    // stage 64x512 z tile as bf16 hi/lo in LDS
    const float* zblk = z + (size_t)blockIdx.x * MTILE * DIMD;
    for (int i = tid; i < MTILE * DIMD / 4; i += 256) {
        float4 v = ((const float4*)zblk)[i];
        int e = i * 4, row = e >> 9, col = e & 511;
        int o = row * LDST + col;
        __bf16 h0 = (__bf16)v.x, h1 = (__bf16)v.y, h2 = (__bf16)v.z, h3 = (__bf16)v.w;
        zhi[o+0] = h0; zhi[o+1] = h1; zhi[o+2] = h2; zhi[o+3] = h3;
        zlo[o+0] = (__bf16)(v.x - (float)h0);
        zlo[o+1] = (__bf16)(v.y - (float)h1);
        zlo[o+2] = (__bf16)(v.z - (float)h2);
        zlo[o+3] = (__bf16)(v.w - (float)h3);
    }
    if (tid < MTILE) bestL[tid] = 0ull;
    __syncthreads();

    unsigned long long best[8];
    #pragma unroll
    for (int r = 0; r < 8; ++r) best[r] = 0ull;

    // A fragment source row for this lane (ISA 16-bit A layout: M = lane&15)
    const __bf16* zhiRow = &zhi[(strip * 16 + m16) * LDST];
    const __bf16* zloRow = &zlo[(strip * 16 + m16) * LDST];

    for (int cbase = 0; cbase < KCB; cbase += NTILE) {
        // 4 code tiles per wave: codes cbase + cg*64 + t*16 + m16
        const int codeB = cbase + cg * 64 + m16;
        const __bf16* bh0 = cb_hi + (size_t)(codeB +  0) * DIMD;
        const __bf16* bh1 = cb_hi + (size_t)(codeB + 16) * DIMD;
        const __bf16* bh2 = cb_hi + (size_t)(codeB + 32) * DIMD;
        const __bf16* bh3 = cb_hi + (size_t)(codeB + 48) * DIMD;
        const __bf16* bl0 = cb_lo + (size_t)(codeB +  0) * DIMD;
        const __bf16* bl1 = cb_lo + (size_t)(codeB + 16) * DIMD;
        const __bf16* bl2 = cb_lo + (size_t)(codeB + 32) * DIMD;
        const __bf16* bl3 = cb_lo + (size_t)(codeB + 48) * DIMD;

        v8f c0 = {}; v8f c1 = {}; v8f c2 = {}; v8f c3 = {};

        #pragma unroll 2
        for (int ks = 0; ks < DIMD; ks += 32) {
            // A (16x32 bf16): lane holds 16 contiguous K at 16*half
            FragB16 ah, al;
            ah.q[0] = *(const uint4*)(zhiRow + ks + 16 * half);
            ah.q[1] = *(const uint4*)(zhiRow + ks + 16 * half + 8);
            al.q[0] = *(const uint4*)(zloRow + ks + 16 * half);
            al.q[1] = *(const uint4*)(zloRow + ks + 16 * half + 8);
            // B (32x16 bf16): lane holds 16 contiguous K at 16*half
            const int bo = ks + 16 * half;
            FragB16 f0h, f1h, f2h, f3h, f0l, f1l, f2l, f3l;
            f0h.q[0] = *(const uint4*)(bh0 + bo); f0h.q[1] = *(const uint4*)(bh0 + bo + 8);
            f1h.q[0] = *(const uint4*)(bh1 + bo); f1h.q[1] = *(const uint4*)(bh1 + bo + 8);
            f2h.q[0] = *(const uint4*)(bh2 + bo); f2h.q[1] = *(const uint4*)(bh2 + bo + 8);
            f3h.q[0] = *(const uint4*)(bh3 + bo); f3h.q[1] = *(const uint4*)(bh3 + bo + 8);
            f0l.q[0] = *(const uint4*)(bl0 + bo); f0l.q[1] = *(const uint4*)(bl0 + bo + 8);
            f1l.q[0] = *(const uint4*)(bl1 + bo); f1l.q[1] = *(const uint4*)(bl1 + bo + 8);
            f2l.q[0] = *(const uint4*)(bl2 + bo); f2l.q[1] = *(const uint4*)(bl2 + bo + 8);
            f3l.q[0] = *(const uint4*)(bl3 + bo); f3l.q[1] = *(const uint4*)(bl3 + bo + 8);

            // 4 independent accumulation chains, 3 WMMAs each (bf16x3 ~ fp32)
            c0 = __builtin_amdgcn_wmma_f32_16x16x32_bf16(false, ah.v, false, f0h.v, (short)0, c0, false, false);
            c1 = __builtin_amdgcn_wmma_f32_16x16x32_bf16(false, ah.v, false, f1h.v, (short)0, c1, false, false);
            c2 = __builtin_amdgcn_wmma_f32_16x16x32_bf16(false, ah.v, false, f2h.v, (short)0, c2, false, false);
            c3 = __builtin_amdgcn_wmma_f32_16x16x32_bf16(false, ah.v, false, f3h.v, (short)0, c3, false, false);
            c0 = __builtin_amdgcn_wmma_f32_16x16x32_bf16(false, ah.v, false, f0l.v, (short)0, c0, false, false);
            c1 = __builtin_amdgcn_wmma_f32_16x16x32_bf16(false, ah.v, false, f1l.v, (short)0, c1, false, false);
            c2 = __builtin_amdgcn_wmma_f32_16x16x32_bf16(false, ah.v, false, f2l.v, (short)0, c2, false, false);
            c3 = __builtin_amdgcn_wmma_f32_16x16x32_bf16(false, ah.v, false, f3l.v, (short)0, c3, false, false);
            c0 = __builtin_amdgcn_wmma_f32_16x16x32_bf16(false, al.v, false, f0h.v, (short)0, c0, false, false);
            c1 = __builtin_amdgcn_wmma_f32_16x16x32_bf16(false, al.v, false, f1h.v, (short)0, c1, false, false);
            c2 = __builtin_amdgcn_wmma_f32_16x16x32_bf16(false, al.v, false, f2h.v, (short)0, c2, false, false);
            c3 = __builtin_amdgcn_wmma_f32_16x16x32_bf16(false, al.v, false, f3h.v, (short)0, c3, false, false);
        }

        // score = 2*dot - ||e||^2 ; fold into per-lane running argmax keys
        const float n0 = norms[codeB +  0];
        const float n1 = norms[codeB + 16];
        const float n2 = norms[codeB + 32];
        const float n3 = norms[codeB + 48];
        #pragma unroll
        for (int r = 0; r < 8; ++r) {
            unsigned long long k0 = packKey(2.0f * c0[r] - n0, codeB +  0);
            unsigned long long k1 = packKey(2.0f * c1[r] - n1, codeB + 16);
            unsigned long long k2 = packKey(2.0f * c2[r] - n2, codeB + 32);
            unsigned long long k3 = packKey(2.0f * c3[r] - n3, codeB + 48);
            if (k0 > best[r]) best[r] = k0;
            if (k1 > best[r]) best[r] = k1;
            if (k2 > best[r]) best[r] = k2;
            if (k3 > best[r]) best[r] = k3;
        }
    }

    // reduce across the 16 lanes of each half-wave (same C-row, different codes)
    #pragma unroll
    for (int r = 0; r < 8; ++r) {
        unsigned long long b = best[r];
        #pragma unroll
        for (int m = 1; m < 16; m <<= 1) {
            unsigned long long o = shflXor64(b, m);
            if (o > b) b = o;
        }
        if (m16 == 0) {
            int tileRow = strip * 16 + r + 8 * half;   // C layout: row = r + 8*half
            atomicMax(&bestL[tileRow], b);             // combine the 2 code-group waves
        }
    }
    __syncthreads();

    if (tid < MTILE) {
        unsigned long long b = bestL[tid];
        int code = 8191 - (int)(unsigned)(b & 0xFFFFFFFFull);
        idx_out[(size_t)blockIdx.x * MTILE + tid] = (float)code;
    }
}

// ---- gather quantized vectors + MSE partial sums ---------------------------
__global__ __launch_bounds__(128)
void gather_loss_kernel(const float* __restrict__ z,
                        const float* __restrict__ cb,
                        const float* __restrict__ idxf,
                        float* __restrict__ outq,
                        float* __restrict__ acc) {
    const int n = blockIdx.x;
    const int code = (int)idxf[n];
    const float4* q4 = (const float4*)(cb + (size_t)code * DIMD);
    const float4* z4 = (const float4*)(z + (size_t)n * DIMD);
    float4* o4 = (float4*)(outq + (size_t)n * DIMD);

    int j = threadIdx.x;                 // 128 threads x 1 float4 = 512 floats
    float4 q = q4[j], zz = z4[j];
    o4[j] = q;                            // forward value of straight-through = quantized
    float dx = zz.x - q.x, dy = zz.y - q.y, dz = zz.z - q.z, dw = zz.w - q.w;
    float s = dx * dx + dy * dy + dz * dz + dw * dw;

    #pragma unroll
    for (int m = 16; m >= 1; m >>= 1) s += __shfl_xor(s, m, 32);
    __shared__ float wsum[4];
    if ((threadIdx.x & 31) == 0) wsum[threadIdx.x >> 5] = s;
    __syncthreads();
    if (threadIdx.x == 0) atomicAdd(acc, wsum[0] + wsum[1] + wsum[2] + wsum[3]);
}

__global__ void finalize_kernel(const float* __restrict__ acc, float* __restrict__ lossOut) {
    // loss = codebook_loss + 0.25 * commitment_loss = 1.25 * mean((z-q)^2)
    *lossOut = acc[0] * 1.25f / (float)(NTOK * DIMD);
}

// ---------------------------------------------------------------------------
extern "C" void kernel_launch(void* const* d_in, const int* in_sizes, int n_in,
                              void* d_out, int out_size, void* d_ws, size_t ws_size,
                              hipStream_t stream) {
    (void)in_sizes; (void)n_in; (void)out_size; (void)ws_size;
    const float* z   = (const float*)d_in[0];   // [32,1024,512] fp32
    const float* cbw = (const float*)d_in[1];   // [8192,512] fp32

    float* out     = (float*)d_out;
    float* outq    = out;                           // 16,777,216 floats
    float* outidx  = out + (size_t)NTOK * DIMD;     // 32,768 "floats" (index values)
    float* outloss = outidx + NTOK;                 // 1 float

    // workspace layout (~16.81 MB): cb_hi | cb_lo | norms | loss accumulator
    char* ws = (char*)d_ws;
    __bf16* cb_hi = (__bf16*)ws;
    __bf16* cb_lo = (__bf16*)(ws + (size_t)KCB * DIMD * 2);
    float*  norms = (float*)(ws + (size_t)KCB * DIMD * 4);
    float*  acc   = norms + KCB;

    hipMemsetAsync(acc, 0, sizeof(float), stream);
    cb_split_kernel<<<KCB * DIMD / 1024, 256, 0, stream>>>(cbw, cb_hi, cb_lo);
    cb_norms_kernel<<<KCB / 8, 256, 0, stream>>>(cbw, norms);
    vq_argmin_kernel<<<NTOK / MTILE, 256, 0, stream>>>(z, cb_hi, cb_lo, norms, outidx);
    gather_loss_kernel<<<NTOK, 128, 0, stream>>>(z, cbw, outidx, outq, acc);
    finalize_kernel<<<1, 1, 0, stream>>>(acc, outloss);
}
